// XCA_1700807050104
// MI455X (gfx1250) — matrix operand order repaired and checked by reading, not compiled, not run on hardware
//
#include <hip/hip_runtime.h>
#include <hip/hip_bf16.h>

typedef __attribute__((ext_vector_type(16))) _Float16 v16h;
typedef __attribute__((ext_vector_type(8)))  _Float16 v8h;
typedef __attribute__((ext_vector_type(4)))  _Float16 v4h;
typedef __attribute__((ext_vector_type(8)))  float    v8f;

#define NUM_HEADS 8
#define CCH 384           // channels
#define OC3 1152          // 3*C
#define NTOK 1024         // H*W
#define DH 48             // head dim

static __device__ __forceinline__ v16h ld16_2(const _Float16* p0, const _Float16* p1) {
    v8h lo = *(const v8h*)p0;
    v8h hi = *(const v8h*)p1;
    v16h r;
#pragma unroll
    for (int i = 0; i < 8; ++i) { r[i] = lo[i]; r[i + 8] = hi[i]; }
    return r;
}

static __device__ __forceinline__ v16h ld16_c(const _Float16* p) {
    return *(const v16h*)p;   // 16 contiguous halves (32B)
}

static __device__ __forceinline__ v8f wmma_f16(v16h a, v16h b, v8f c) {
    return __builtin_amdgcn_wmma_f32_16x16x32_f16(false, a, false, b, (short)0, c, false, false);
}

// async copy 16B global -> LDS (gfx1250), tracked by ASYNCcnt
static __device__ __forceinline__ void async_g2l_b128(unsigned lds_off, const void* g) {
    unsigned long long ga = (unsigned long long)g;
    asm volatile("global_load_async_to_lds_b128 %0, %1, off"
                 :: "v"(lds_off), "v"(ga) : "memory");
}
static __device__ __forceinline__ void wait_async0() {
    asm volatile("s_wait_asynccnt 0x0" ::: "memory");
}

// ---------------------------------------------------------------------------
// Kernel 1: QKV = w_qkv(1152x384) * x(b,384,1024) + bias  -> f16 qkv[b][o][n]
// grid: (N/128, OC3/64, B), block 256 (8 waves)
// ---------------------------------------------------------------------------
#define BM 64
#define BN 128
#define BK 32
#define APAD 8   // row stride BK+8 = 40 halves = 80B (16B aligned)

__global__ __launch_bounds__(256)
void xca_qkv_gemm(const float* __restrict__ w, const float* __restrict__ bias,
                  const float* __restrict__ x, _Float16* __restrict__ qkv) {
    __shared__ _Float16 As[BM][BK + APAD];
    __shared__ _Float16 Bt[BN][BK + APAD];

    const int tid  = threadIdx.x;
    const int lane = tid & 31;
    const int wid  = tid >> 5;
    const int b    = blockIdx.z;
    const int Mb   = blockIdx.y * BM;
    const int Nb   = blockIdx.x * BN;

    const int row = lane & 15;
    const int hi  = lane >> 4;
    const int Moff = (wid >> 1) * 16;
    const int Noff = (wid & 1) * 64;

    // staging coordinates (compile-time friendly)
    const int am  = tid >> 2;            // 0..63  A row
    const int ak8 = (tid & 3) * 8;       // 0,8,16,24

    v8f acc[4];
#pragma unroll
    for (int t = 0; t < 4; ++t)
#pragma unroll
        for (int e = 0; e < 8; ++e) acc[t][e] = 0.0f;

    const size_t xbase = ((size_t)b * CCH) * NTOK;

    for (int k0 = 0; k0 < CCH; k0 += BK) {
        if (k0 + BK < CCH)
            __builtin_prefetch(&w[(size_t)(Mb + am) * CCH + k0 + BK + ak8], 0, 0);

        // ---- stage A (w_qkv f32 -> f16), one 16B-aligned v8h store/thread ----
        {
            const float4* src = (const float4*)&w[(size_t)(Mb + am) * CCH + k0 + ak8];
            float4 f0 = src[0], f1 = src[1];
            v8h hp;
            hp[0] = (_Float16)f0.x; hp[1] = (_Float16)f0.y;
            hp[2] = (_Float16)f0.z; hp[3] = (_Float16)f0.w;
            hp[4] = (_Float16)f1.x; hp[5] = (_Float16)f1.y;
            hp[6] = (_Float16)f1.z; hp[7] = (_Float16)f1.w;
            *(v8h*)&As[am][ak8] = hp;
        }
        // ---- stage B transposed: Bt[n][k] = x[b,k0+k,Nb+n]; pack 4 k / store ----
#pragma unroll
        for (int p = 0; p < 4; ++p) {
            int i = tid + p * 256;       // 0..1023
            int n = i & 127;
            int k = (i >> 7) * 4;        // 0,4,...,28
            v4h pk;
#pragma unroll
            for (int j = 0; j < 4; ++j)
                pk[j] = (_Float16)x[xbase + (size_t)(k0 + k + j) * NTOK + Nb + n];
            *(v4h*)&Bt[n][k] = pk;
        }
        __syncthreads();

        const int kbA = hi * 16;
        v16h afrag = ld16_2(&As[Moff + row][kbA], &As[Moff + row][kbA + 8]);
        const int kbB = hi * 16;
#pragma unroll
        for (int t = 0; t < 4; ++t) {
            v16h bfrag = ld16_c(&Bt[Noff + 16 * t + row][kbB]);
            acc[t] = wmma_f16(afrag, bfrag, acc[t]);
        }
        __syncthreads();
    }

#pragma unroll
    for (int t = 0; t < 4; ++t) {
        const int nG = Nb + Noff + 16 * t + row;
#pragma unroll
        for (int r = 0; r < 8; ++r) {
            const int o = Mb + Moff + r + hi * 8;
            float val = acc[t][r] + bias[o];
            qkv[((size_t)b * OC3 + o) * NTOK + nG] = (_Float16)val;
        }
    }
}

// ---------------------------------------------------------------------------
// Kernel 2: per-(b,h) XCA core. Writes attention output TRANSPOSED:
//   aoutT[b][n][c]  (c = h*48 + d), enabling 16B stores here and pure
//   byte-copy (async-to-LDS) staging in the projection GEMM.
// grid = B*NUM_HEADS, block 288 (9 waves).
// ---------------------------------------------------------------------------
__global__ __launch_bounds__(288)
void xca_attn(const _Float16* __restrict__ qkv, const float* __restrict__ temperature,
              _Float16* __restrict__ aoutT) {
    __shared__ float    sQ[DH];
    __shared__ float    sK[DH];
    __shared__ float    attnF[DH][DH];
    __shared__ _Float16 attnH[DH][64];     // K-padded to 64
    __shared__ _Float16 Vt[256][64];       // v^T chunk, e-padded to 64

    const int tid  = threadIdx.x;
    const int lane = tid & 31;
    const int w    = tid >> 5;             // 0..8
    const int bh   = blockIdx.x;
    const int b    = bh >> 3;
    const int h    = bh & 7;

    const _Float16* qh = qkv + ((size_t)b * OC3 + h * DH) * NTOK;
    const _Float16* kh = qkv + ((size_t)b * OC3 + CCH + h * DH) * NTOK;
    const _Float16* vh = qkv + ((size_t)b * OC3 + 2 * CCH + h * DH) * NTOK;

    // ---- inverse l2 norms; one wave/row, v8h loads, wave32 shuffle reduce ----
    for (int rrow = w; rrow < 2 * DH; rrow += 9) {
        const _Float16* p = (rrow < DH) ? (qh + (size_t)rrow * NTOK)
                                        : (kh + (size_t)(rrow - DH) * NTOK);
        float s = 0.0f;
        for (int nb = 0; nb < NTOK; nb += 256) {
            v8h vv = *(const v8h*)(p + nb + lane * 8);
#pragma unroll
            for (int j = 0; j < 8; ++j) { float f = (float)vv[j]; s += f * f; }
        }
#pragma unroll
        for (int off = 16; off > 0; off >>= 1) s += __shfl_xor(s, off, 32);
        if (lane == 0) {
            float sc = 1.0f / fmaxf(sqrtf(s), 1e-12f);
            if (rrow < DH) sQ[rrow] = sc; else sK[rrow - DH] = sc;
        }
    }
    // zero attnH pad and Vt pad columns (48..63) once
    for (int i = tid; i < DH * 64; i += 288)
        (&attnH[0][0])[i] = (_Float16)0.0f;
    {
        v8h z; 
#pragma unroll
        for (int j = 0; j < 8; ++j) z[j] = (_Float16)0.0f;
        for (int i = tid; i < 512; i += 288) {
            int n = i >> 1, gp = i & 1;
            *(v8h*)&Vt[n][48 + 8 * gp] = z;
        }
    }
    __syncthreads();

    // ---- Gram: raw[d,e] = sum_n q[d,n]*k[e,n]; 3x3 tiles over 9 waves ----
    const int dt = w / 3, et = w % 3;
    const int Doff = dt * 16, Eoff = et * 16;
    const int row = lane & 15;
    const int hi  = lane >> 4;
    {
        v8f acc;
#pragma unroll
        for (int e = 0; e < 8; ++e) acc[e] = 0.0f;
        const int kbA = hi * 16, kbB = hi * 16;
        for (int nb = 0; nb < NTOK; nb += 32) {
            const _Float16* ap = qh + (size_t)(Doff + row) * NTOK + nb + kbA;
            v16h afrag = ld16_2(ap, ap + 8);
            v16h bfrag = ld16_c(kh + (size_t)(Eoff + row) * NTOK + nb + kbB);
            acc = wmma_f16(afrag, bfrag, acc);
        }
        const float tmp = temperature[h];
        const int e = Eoff + row;
#pragma unroll
        for (int r = 0; r < 8; ++r) {
            const int d = Doff + r + hi * 8;
            attnF[d][e] = acc[r] * sQ[d] * sK[e] * tmp;
        }
    }
    __syncthreads();

    // ---- softmax over e (48) per row d ----
    if (tid < DH) {
        float m = -3.4e38f;
#pragma unroll 4
        for (int e = 0; e < DH; ++e) m = fmaxf(m, attnF[tid][e]);
        float sum = 0.0f;
#pragma unroll 4
        for (int e = 0; e < DH; ++e) sum += __expf(attnF[tid][e] - m);
        float inv = 1.0f / sum;
#pragma unroll 4
        for (int e = 0; e < DH; ++e)
            attnH[tid][e] = (_Float16)(__expf(attnF[tid][e] - m) * inv);
    }
    __syncthreads();

    // ---- out[d,n] = attn(48x48) @ v(48xN); chunk N by 256 via LDS v^T ----
    const int wn = w % 3;
    for (int c0 = 0; c0 < NTOK; c0 += 256) {
        // transpose-stage v chunk: pack 8 e-values per 16B LDS store
        for (int i = tid; i < 256 * 6; i += 288) {
            int g = i / 256;           // 0..5  -> e block of 8
            int n = i & 255;
            v8h pk;
#pragma unroll
            for (int j = 0; j < 8; ++j)
                pk[j] = vh[(size_t)(g * 8 + j) * NTOK + c0 + n];
            *(v8h*)&Vt[n][g * 8] = pk;
        }
        __syncthreads();

        for (int j = wn; j < 16; j += 3) {
            v8f acc;
#pragma unroll
            for (int e = 0; e < 8; ++e) acc[e] = 0.0f;
#pragma unroll
            for (int ks = 0; ks < 2; ++ks) {
                const int eb = ks * 32;
                const int kbA = hi * 16, kbB = hi * 16;
                v16h afrag = ld16_2(&attnH[Doff + row][eb + kbA],
                                    &attnH[Doff + row][eb + kbA + 8]);
                v16h bfrag = ld16_c(&Vt[16 * j + row][eb + kbB]);
                acc = wmma_f16(afrag, bfrag, acc);
            }
            // transposed store: 8 consecutive channels per lane -> one 16B store
            const int n  = c0 + 16 * j + row;
            const int cb = h * DH + Doff + hi * 8;
            v8h hv;
#pragma unroll
            for (int r = 0; r < 8; ++r) hv[r] = (_Float16)acc[r];
            *(v8h*)&aoutT[((size_t)b * NTOK + n) * CCH + cb] = hv;
        }
        __syncthreads();
    }
}

// ---------------------------------------------------------------------------
// Kernel 3: out = x + w_proj(384x384) @ aoutT^T + b_proj  (f32 output)
// B tile is a pure byte copy from aoutT[b][n][c] -> staged with
// global_load_async_to_lds_b128 (ASYNCcnt path).
// grid: (N/128, C/64, B), block 256
// ---------------------------------------------------------------------------
__global__ __launch_bounds__(256)
void xca_proj(const float* __restrict__ w, const float* __restrict__ bias,
              const _Float16* __restrict__ aT, const float* __restrict__ x,
              float* __restrict__ out) {
    __shared__ _Float16 As[BM][BK + APAD];
    __shared__ _Float16 Bt[BN][BK + APAD];

    const int tid  = threadIdx.x;
    const int lane = tid & 31;
    const int wid  = tid >> 5;
    const int b    = blockIdx.z;
    const int Mb   = blockIdx.y * BM;
    const int Nb   = blockIdx.x * BN;

    const int row = lane & 15;
    const int hi  = lane >> 4;
    const int Moff = (wid >> 1) * 16;
    const int Noff = (wid & 1) * 64;

    const int am  = tid >> 2;
    const int ak8 = (tid & 3) * 8;

    v8f acc[4];
#pragma unroll
    for (int t = 0; t < 4; ++t)
#pragma unroll
        for (int e = 0; e < 8; ++e) acc[t][e] = 0.0f;

    for (int k0 = 0; k0 < CCH; k0 += BK) {
        // ---- async B tile: Bt[n][k0..k0+31] = aoutT[b][Nb+n][k0..]; 2x16B/thread ----
#pragma unroll
        for (int p = 0; p < 2; ++p) {
            int c  = tid + p * 256;      // 0..511
            int n  = c >> 2;
            int ch = c & 3;              // 16B chunk within the 64B row
            const _Float16* g = aT + ((size_t)b * NTOK + Nb + n) * CCH + k0 + ch * 8;
            async_g2l_b128((unsigned)(unsigned long long)&Bt[n][ch * 8], g);
        }
        if (k0 + BK < CCH)
            __builtin_prefetch(&w[(size_t)(Mb + am) * CCH + k0 + BK + ak8], 0, 0);
        // ---- stage A (w_proj f32 -> f16) ----
        {
            const float4* src = (const float4*)&w[(size_t)(Mb + am) * CCH + k0 + ak8];
            float4 f0 = src[0], f1 = src[1];
            v8h hp;
            hp[0] = (_Float16)f0.x; hp[1] = (_Float16)f0.y;
            hp[2] = (_Float16)f0.z; hp[3] = (_Float16)f0.w;
            hp[4] = (_Float16)f1.x; hp[5] = (_Float16)f1.y;
            hp[6] = (_Float16)f1.z; hp[7] = (_Float16)f1.w;
            *(v8h*)&As[am][ak8] = hp;
        }
        wait_async0();
        __syncthreads();

        const int kbA = hi * 16;
        v16h afrag = ld16_2(&As[Moff + row][kbA], &As[Moff + row][kbA + 8]);
        const int kbB = hi * 16;
#pragma unroll
        for (int t = 0; t < 4; ++t) {
            v16h bfrag = ld16_c(&Bt[Noff + 16 * t + row][kbB]);
            acc[t] = wmma_f16(afrag, bfrag, acc[t]);
        }
        __syncthreads();
    }

#pragma unroll
    for (int t = 0; t < 4; ++t) {
        const int nG = Nb + Noff + 16 * t + row;
#pragma unroll
        for (int r = 0; r < 8; ++r) {
            const int o = Mb + Moff + r + hi * 8;
            const size_t idx = ((size_t)b * CCH + o) * NTOK + nG;
            out[idx] = acc[t][r] + bias[o] + x[idx];
        }
    }
}

// ---------------------------------------------------------------------------
extern "C" void kernel_launch(void* const* d_in, const int* in_sizes, int n_in,
                              void* d_out, int out_size, void* d_ws, size_t ws_size,
                              hipStream_t stream) {
    const float* x      = (const float*)d_in[0];
    const float* w_qkv  = (const float*)d_in[1];
    const float* b_qkv  = (const float*)d_in[2];
    const float* temp   = (const float*)d_in[3];
    const float* w_proj = (const float*)d_in[4];
    const float* b_proj = (const float*)d_in[5];
    float* out = (float*)d_out;

    const int B = 32;
    _Float16* qkv  = (_Float16*)d_ws;
    size_t qkv_bytes = (size_t)B * OC3 * NTOK * sizeof(_Float16);
    qkv_bytes = (qkv_bytes + 255) & ~(size_t)255;
    _Float16* aoutT = (_Float16*)((char*)d_ws + qkv_bytes);

    {
        dim3 grid(NTOK / BN, OC3 / BM, B);
        xca_qkv_gemm<<<grid, 256, 0, stream>>>(w_qkv, b_qkv, x, qkv);
    }
    {
        dim3 grid(B * NUM_HEADS);
        xca_attn<<<grid, 288, 0, stream>>>(qkv, temp, aoutT);
    }
    {
        dim3 grid(NTOK / BN, CCH / BM, B);
        xca_proj<<<grid, 256, 0, stream>>>(w_proj, b_proj, aoutT, x, out);
    }
}
